// LM_48670569398641
// MI455X (gfx1250) — matrix operand order repaired and checked by reading, not compiled, not run on hardware
//
#include <hip/hip_runtime.h>
#include <hip/hip_bf16.h>
#include <stdint.h>

// Problem constants (from reference): V,E,L,B,T,D
#define V_  32000
#define E_  512
#define L_  2
#define B_  8
#define T_  128
#define D_  4
#define TN_ (T_ - D_)      // 124
#define P_ENC (B_ * T_)    // 1024
#define P_DEC (B_ * TN_)   // 992
#define G3E  (3 * E_)      // 1536

typedef __bf16 bf16_t;
typedef __attribute__((ext_vector_type(16))) __bf16 v16bf;
typedef __attribute__((ext_vector_type(8)))  __bf16 v8bf;
typedef __attribute__((ext_vector_type(8)))  float  v8f;
typedef __attribute__((ext_vector_type(4)))  float  v4f;

__device__ __forceinline__ bf16_t f2bf(float f) {
  // round-to-nearest-even f32 -> bf16
  uint32_t u = __builtin_bit_cast(uint32_t, f);
  uint32_t r = (u + 0x7FFFu + ((u >> 16) & 1u)) >> 16;
  return __builtin_bit_cast(bf16_t, (uint16_t)r);
}

// ---------------------------------------------------------------------------
// bf16 WMMA GEMM:  C[M,N] = A[M,K] @ W^T (+ bias), W given as W[N][K] row-major.
//   GATHER=1 : A row m -> embedded[b, (t-1+d) mod T]  (decoder shifted inputs)
//   MODE=0   : f32 output + bias, row remap m*rowMul+rowAdd,
//              LDS-staged fully-coalesced float4 write-out
//   MODE=1   : bf16 output, no bias (direct stores; tiny GEMM only)
//
// Block: 256 threads = 8 waves. Block tile 64(M) x 256(N), K-chunk 32.
// Wave grid 2x4; each wave owns a 32x64 tile = 8 v_wmma_f32_16x16x32_bf16/chunk.
// LDS staging writes contiguous-k fragments (lane half h: k = 16h + s) with contiguous
// b128 runs; fragment reads are 2x ds_load_b128 per 16x16 operand.
// Global loads for chunk k+1 are issued before the chunk-k WMMAs (register
// pipeline); global_prefetch_b8 warms chunk k+2 of B.
// ---------------------------------------------------------------------------
template<int GATHER, int MODE>
__global__ __launch_bounds__(256)
void gemm_kern(const bf16_t* __restrict__ A, const bf16_t* __restrict__ Bm,
               float* __restrict__ Cf, bf16_t* __restrict__ Cb,
               const float* __restrict__ bias,
               int M, int K, int lda, int ldb, int ldc,
               long long aBatchStride, long long bBatchStride, long long cBatchStride,
               int rowMul, int rowAdd, int gTn, int gT, int gD)
{
  // MODE 0: reuse the 20KB staging region as a 64KB f32 block tile at epilogue
  __shared__ __align__(32) uint8_t smem[(MODE == 0) ? 65536 : 20480];
  bf16_t (*Af)[32][16] = (bf16_t (*)[32][16])(smem);          //  4 M-tiles (4 KB)
  bf16_t (*Bf)[32][16] = (bf16_t (*)[32][16])(smem + 4096);   // 16 N-tiles (16 KB)

  const int tid = threadIdx.x;
  const int bz  = blockIdx.z;
  A  += (size_t)bz * aBatchStride;
  Bm += (size_t)bz * bBatchStride;
  if (MODE == 0) Cf += (size_t)bz * cBatchStride;
  else           Cb += (size_t)bz * cBatchStride;

  const int n0 = blockIdx.x * 256;
  const int m0 = blockIdx.y * 64;

  const int lane = tid & 31;
  const int wid  = tid >> 5;
  const int wm   = wid >> 2;   // 0..1 : wave M position
  const int wn   = wid & 3;    // 0..3 : wave N position

  // ---- loop-invariant staging addresses ----
  // A: thread t<128 owns local row ml = t>>1, k sub-half kh = (t&1)*16
  const int ml = tid >> 1;
  const int kh = (tid & 1) << 4;
  const bf16_t* aSrc = A;
  bool aValid = false;
  if (tid < 128) {
    const int mg = m0 + ml;
    aValid = (mg < M);
    int row = aValid ? mg : 0;
    if (GATHER && aValid) {
      const int b = mg / gTn, t = mg - b * gTn;
      row = b * gT + ((t - 1 + gD + gT) % gT);
    }
    aSrc = A + (size_t)row * lda + kh;
  }
  // B: thread owns col nl = tid (256 cols of W[N][K])
  const int nl = tid;
  const bf16_t* bSrc = Bm + (size_t)(n0 + nl) * ldb;

  // LDS staging destinations (loop-invariant)
  bf16_t* aDst0 = &Af[ml >> 4][(ml & 15) | kh][0];
  bf16_t* aDst1 = &Af[ml >> 4][(ml & 15) | kh][8];
  bf16_t* bDst0 = &Bf[nl >> 4][(nl & 15)     ][0];
  bf16_t* bDst1 = &Bf[nl >> 4][(nl & 15) | 16][0];

  v8f acc[2][4] = {};
  v8bf  ar0 = {}, ar1 = {};
  v16bf br0, br1;

  const int nk = K >> 5;       // K is 128 or 512

  // ---- pipeline prologue: load chunk 0 into registers ----
  if (aValid) {
    ar0 = *(const v8bf*)(aSrc);
    ar1 = *(const v8bf*)(aSrc + 8);
  }
  br0 = *(const v16bf*)(bSrc);
  br1 = *(const v16bf*)(bSrc + 16);

  for (int kc = 0; kc < nk; ++kc) {
    __syncthreads();           // previous chunk's fragment reads done
    // ---- stage registers -> LDS fragments (contiguous b128 runs) ----
    if (tid < 128) {
      *(v8bf*)aDst0 = ar0;     // k = kh+0..7  (half 0)
      *(v8bf*)aDst1 = ar1;     // k = kh+8..15 (half 1)
    }
    *(v16bf*)bDst0 = br0;      // k 0..15  at lane n&15
    *(v16bf*)bDst1 = br1;      // k 16..31 at lane 16|(n&15)
    __syncthreads();

    // ---- issue global loads for chunk k+1 (overlap with WMMAs below) ----
    if (kc + 1 < nk) {
      const int k0n = (kc + 1) << 5;
      if (aValid) {
        ar0 = *(const v8bf*)(aSrc + k0n);
        ar1 = *(const v8bf*)(aSrc + k0n + 8);
      }
      br0 = *(const v16bf*)(bSrc + k0n);
      br1 = *(const v16bf*)(bSrc + k0n + 16);
    }
    if (kc + 2 < nk)           // warm chunk k+2 of B (global_prefetch_b8)
      __builtin_prefetch(bSrc + ((kc + 2) << 5), 0, 1);

    // ---- load all fragments, then 8 back-to-back WMMAs (32x64 wave tile) ----
    v16bf a0 = *(const v16bf*)&Af[wm * 2 + 0][lane][0];
    v16bf a1 = *(const v16bf*)&Af[wm * 2 + 1][lane][0];
    v16bf b0 = *(const v16bf*)&Bf[wn * 4 + 0][lane][0];
    v16bf b1 = *(const v16bf*)&Bf[wn * 4 + 1][lane][0];
    v16bf b2 = *(const v16bf*)&Bf[wn * 4 + 2][lane][0];
    v16bf b3 = *(const v16bf*)&Bf[wn * 4 + 3][lane][0];

    acc[0][0] = __builtin_amdgcn_wmma_f32_16x16x32_bf16(false, a0, false, b0, (short)0, acc[0][0], false, false);
    acc[1][0] = __builtin_amdgcn_wmma_f32_16x16x32_bf16(false, a1, false, b0, (short)0, acc[1][0], false, false);
    acc[0][1] = __builtin_amdgcn_wmma_f32_16x16x32_bf16(false, a0, false, b1, (short)0, acc[0][1], false, false);
    acc[1][1] = __builtin_amdgcn_wmma_f32_16x16x32_bf16(false, a1, false, b1, (short)0, acc[1][1], false, false);
    acc[0][2] = __builtin_amdgcn_wmma_f32_16x16x32_bf16(false, a0, false, b2, (short)0, acc[0][2], false, false);
    acc[1][2] = __builtin_amdgcn_wmma_f32_16x16x32_bf16(false, a1, false, b2, (short)0, acc[1][2], false, false);
    acc[0][3] = __builtin_amdgcn_wmma_f32_16x16x32_bf16(false, a0, false, b3, (short)0, acc[0][3], false, false);
    acc[1][3] = __builtin_amdgcn_wmma_f32_16x16x32_bf16(false, a1, false, b3, (short)0, acc[1][3], false, false);
  }

  // ---- epilogue: C/D layout lanes 0-15: M=r, lanes 16-31: M=8+r, N=lane&15 ----
  const int half = lane >> 4;
  const int col  = lane & 15;

  if (MODE == 0) {
    // stage f32 block tile in LDS, then fully-coalesced float4 write-out
    __syncthreads();                       // staging buffer reads done
    float* Cs = (float*)smem;              // [64][256]
#pragma unroll
    for (int i = 0; i < 2; ++i) {
#pragma unroll
      for (int j = 0; j < 4; ++j) {
        const int crow = wm * 32 + i * 16 + half * 8;
        const int ccol = wn * 64 + j * 16 + col;
        const float bv = bias[n0 + ccol];
#pragma unroll
        for (int r = 0; r < 8; ++r)
          Cs[(crow + r) * 256 + ccol] = acc[i][j][r] + bv;
      }
    }
    __syncthreads();
    const int c4 = (tid & 63) << 2;        // 64 threads cover one 1KB row
#pragma unroll
    for (int r = 0; r < 16; ++r) {
      const int row = (tid >> 6) + r * 4;
      const int m = m0 + row;
      if (m < M) {
        v4f v = *(const v4f*)&Cs[row * 256 + c4];
        *(v4f*)&Cf[(size_t)(m * rowMul + rowAdd) * ldc + n0 + c4] = v;
      }
    }
  } else {
    // bf16 direct stores (used only by the small einsum GEMM)
#pragma unroll
    for (int i = 0; i < 2; ++i) {
#pragma unroll
      for (int j = 0; j < 4; ++j) {
        const int mt = m0 + (wm * 2 + i) * 16 + half * 8;
        const int nt = n0 + (wn * 4 + j) * 16 + col;
#pragma unroll
        for (int r = 0; r < 8; ++r) {
          const int m = mt + r;
          if (m < M) Cb[(size_t)m * ldc + nt] = f2bf(acc[i][j][r]);
        }
      }
    }
  }
}

// ---------------------------------------------------------------------------
// Elementwise kernels
// ---------------------------------------------------------------------------
__global__ void embed_kernel(const int* __restrict__ x, const float* __restrict__ emb,
                             float* __restrict__ embF, bf16_t* __restrict__ embB,
                             float* __restrict__ infF, bf16_t* __restrict__ infB)
{
  const int g = blockIdx.x * blockDim.x + threadIdx.x;
  if (g >= B_ * T_ * E_) return;
  const int e  = g & (E_ - 1);
  const int bt = g >> 9;                 // E_ == 512
  const int tok = x[bt];
  const float v = (tok == 0) ? 0.f : emb[(size_t)tok * E_ + e];
  const bf16_t b = f2bf(v);
  embF[g] = v; embB[g] = b;
  infF[g] = v; infB[g] = b;
}

__global__ void f32_to_bf16(const float* __restrict__ in, bf16_t* __restrict__ out, int n)
{
  const int g = blockIdx.x * blockDim.x + threadIdx.x;
  if (g < n) out[g] = f2bf(in[g]);
}

// xT[b][e][t] = in[b][t][e]  (bf16) so the einsum's X operand is in W[N][K] form
__global__ void transpose_bf16(const float* __restrict__ in, bf16_t* __restrict__ outT)
{
  const int g = blockIdx.x * blockDim.x + threadIdx.x;
  if (g >= B_ * T_ * E_) return;
  const int t = g & (T_ - 1);
  const int e = (g >> 7) & (E_ - 1);
  const int b = g >> 16;                 // T_*E_ == 65536
  outT[g] = f2bf(in[((size_t)b * T_ + t) * E_ + e]);
}

// Agl[b][i][j] = G[b][l][j][i]  (so the einsum becomes plain A @ X)
__global__ void gprep_kernel(const float* __restrict__ G, bf16_t* __restrict__ Agl, int l)
{
  const int g = blockIdx.x * blockDim.x + threadIdx.x;
  if (g >= B_ * T_ * T_) return;
  const int j = g & (T_ - 1);
  const int i = (g >> 7) & (T_ - 1);
  const int b = g >> 14;
  Agl[g] = f2bf(G[(((size_t)b * L_ + l) * T_ + j) * T_ + i]);
}

__global__ void gru_combine(const float* __restrict__ gi, const float* __restrict__ gh,
                            const float* __restrict__ hprev,
                            float* __restrict__ hF, bf16_t* __restrict__ hB, int P)
{
  const int g = blockIdx.x * blockDim.x + threadIdx.x;
  if (g >= P * E_) return;
  const int e = g & (E_ - 1);
  const int p = g >> 9;
  const size_t base = (size_t)p * G3E;
  const float ir = gi[base + e], iz = gi[base + E_ + e], in_ = gi[base + 2 * E_ + e];
  const float hr = gh[base + e], hz = gh[base + E_ + e], hn  = gh[base + 2 * E_ + e];
  const float r = 1.f / (1.f + expf(-(ir + hr)));
  const float z = 1.f / (1.f + expf(-(iz + hz)));
  const float n = tanhf(in_ + r * hn);
  const float h = hprev[g];
  const float o = (1.f - z) * n + z * h;
  hF[g] = o; hB[g] = f2bf(o);
}

__global__ void h0_init(const float* __restrict__ infF,
                        float* __restrict__ hF, bf16_t* __restrict__ hB)
{
  const int g = blockIdx.x * blockDim.x + threadIdx.x;
  if (g >= P_DEC * E_) return;
  const int e = g & (E_ - 1);
  const int p = g >> 9;
  const int b = p / TN_, t = p - b * TN_;
  const float v = infF[((size_t)(b * T_ + t)) * E_ + e];
  hF[g] = v; hB[g] = f2bf(v);
}

// ---------------------------------------------------------------------------
// Host launch
// ---------------------------------------------------------------------------
extern "C" void kernel_launch(void* const* d_in, const int* in_sizes, int n_in,
                              void* d_out, int out_size, void* d_ws, size_t ws_size,
                              hipStream_t stream)
{
  (void)in_sizes; (void)n_in; (void)out_size; (void)ws_size;
  const int*   x     = (const int*)  d_in[0];
  // d_in[1] = lengths : unused by the reference
  const float* G     = (const float*)d_in[2];
  const float* emb   = (const float*)d_in[3];
  const float* encWi = (const float*)d_in[4];
  const float* encWh = (const float*)d_in[5];
  const float* encBi = (const float*)d_in[6];
  const float* encBh = (const float*)d_in[7];
  const float* decWi = (const float*)d_in[8];
  const float* decWh = (const float*)d_in[9];
  const float* decBi = (const float*)d_in[10];
  const float* decBh = (const float*)d_in[11];
  const float* ffW   = (const float*)d_in[12];
  const float* ffB   = (const float*)d_in[13];
  float* out = (float*)d_out;

  uint8_t* ws = (uint8_t*)d_ws;
  size_t off = 0;
  auto alloc = [&](size_t bytes) -> void* {
    void* p = ws + off;
    off = (off + bytes + 255) & ~(size_t)255;
    return p;
  };

  float*  embF  = (float*) alloc((size_t)P_ENC * E_ * 4);
  bf16_t* embB  = (bf16_t*)alloc((size_t)P_ENC * E_ * 2);
  float*  infF  = (float*) alloc((size_t)P_ENC * E_ * 4);
  bf16_t* infB  = (bf16_t*)alloc((size_t)P_ENC * E_ * 2);
  bf16_t* xTB   = (bf16_t*)alloc((size_t)P_ENC * E_ * 2);   // input_f^T per layer
  bf16_t* wgtB  = (bf16_t*)alloc((size_t)P_ENC * E_ * 2);
  float*  giBuf = (float*) alloc((size_t)P_ENC * G3E * 4);
  float*  ghBuf = (float*) alloc((size_t)P_ENC * G3E * 4);
  float*  hF    = (float*) alloc((size_t)P_DEC * E_ * 4);
  bf16_t* hB    = (bf16_t*)alloc((size_t)P_DEC * E_ * 2);
  bf16_t* AglB  = (bf16_t*)alloc((size_t)B_ * T_ * T_ * 2);
  bf16_t* eWiB  = (bf16_t*)alloc((size_t)G3E * E_ * 2);
  bf16_t* eWhB  = (bf16_t*)alloc((size_t)G3E * E_ * 2);
  bf16_t* dWiB  = (bf16_t*)alloc((size_t)G3E * E_ * 2);
  bf16_t* dWhB  = (bf16_t*)alloc((size_t)G3E * E_ * 2);
  bf16_t* ffWB  = (bf16_t*)alloc((size_t)V_ * E_ * 2);

  const dim3 blk(256);
  auto cdiv = [](int a, int b) { return (a + b - 1) / b; };

  // ---- embedding (token 0 zeroed) ----
  embed_kernel<<<cdiv(P_ENC * E_, 256), blk, 0, stream>>>(x, emb, embF, embB, infF, infB);

  // ---- one-time weight conversions to bf16 ----
  f32_to_bf16<<<cdiv(V_ * E_, 256),  blk, 0, stream>>>(ffW,   ffWB, V_ * E_);
  f32_to_bf16<<<cdiv(G3E * E_, 256), blk, 0, stream>>>(decWi, dWiB, G3E * E_);
  f32_to_bf16<<<cdiv(G3E * E_, 256), blk, 0, stream>>>(decWh, dWhB, G3E * E_);

  // ---- encoder: 2 layers ----
  for (int l = 0; l < L_; ++l) {
    f32_to_bf16<<<cdiv(G3E * E_, 256), blk, 0, stream>>>(encWi + (size_t)l * G3E * E_, eWiB, G3E * E_);
    f32_to_bf16<<<cdiv(G3E * E_, 256), blk, 0, stream>>>(encWh + (size_t)l * G3E * E_, eWhB, G3E * E_);
    gprep_kernel<<<cdiv(B_ * T_ * T_, 256), blk, 0, stream>>>(G, AglB, l);
    transpose_bf16<<<cdiv(P_ENC * E_, 256), blk, 0, stream>>>(infF, xTB);

    // wgt[b] = G[b,l]^T @ input_f[b]  (batched; X supplied transposed; bf16 out)
    gemm_kern<0, 1><<<dim3(E_ / 256, T_ / 64, B_), blk, 0, stream>>>(
        AglB, xTB, nullptr, wgtB, nullptr,
        T_, T_, T_, T_, E_,
        (long long)T_ * T_, (long long)T_ * E_, (long long)T_ * E_,
        1, 0, 0, 0, 0);

    // gi = wgt @ Wi^T + bi ; gh = input_f @ Wh^T + bh
    gemm_kern<0, 0><<<dim3(G3E / 256, cdiv(P_ENC, 64), 1), blk, 0, stream>>>(
        wgtB, eWiB, giBuf, nullptr, encBi + (size_t)l * G3E,
        P_ENC, E_, E_, E_, G3E, 0, 0, 0, 1, 0, 0, 0, 0);
    gemm_kern<0, 0><<<dim3(G3E / 256, cdiv(P_ENC, 64), 1), blk, 0, stream>>>(
        infB, eWhB, ghBuf, nullptr, encBh + (size_t)l * G3E,
        P_ENC, E_, E_, E_, G3E, 0, 0, 0, 1, 0, 0, 0, 0);

    gru_combine<<<cdiv(P_ENC * E_, 256), blk, 0, stream>>>(giBuf, ghBuf, infF, infF, infB, P_ENC);
  }

  // ---- decoder: h0 = input_f[:, :Tn] ----
  h0_init<<<cdiv(P_DEC * E_, 256), blk, 0, stream>>>(infF, hF, hB);

  for (int d = 0; d < D_; ++d) {
    // gi = dec_in_d @ dec_Wi^T + bi  (A rows gathered from embedded)
    gemm_kern<1, 0><<<dim3(G3E / 256, cdiv(P_DEC, 64), 1), blk, 0, stream>>>(
        embB, dWiB, giBuf, nullptr, decBi,
        P_DEC, E_, E_, E_, G3E, 0, 0, 0, 1, 0, TN_, T_, d);
    // gh = h @ dec_Wh^T + bh
    gemm_kern<0, 0><<<dim3(G3E / 256, cdiv(P_DEC, 64), 1), blk, 0, stream>>>(
        hB, dWhB, ghBuf, nullptr, decBh,
        P_DEC, E_, E_, E_, G3E, 0, 0, 0, 1, 0, 0, 0, 0);

    gru_combine<<<cdiv(P_DEC * E_, 256), blk, 0, stream>>>(giBuf, ghBuf, hF, hF, hB, P_DEC);

    // logits = h @ ff_W^T + ff_b  -> out row p goes to p*D + d (interleaved)
    gemm_kern<0, 0><<<dim3(V_ / 256, cdiv(P_DEC, 64), 1), blk, 0, stream>>>(
        hB, ffWB, out, nullptr, ffB,
        P_DEC, E_, E_, E_, V_, 0, 0, 0, D_, d, 0, 0, 0);
  }
}
